// VectorQuantizer_23373212025533
// MI455X (gfx1250) — matrix operand order repaired and checked by reading, not compiled, not run on hardware
//
#include <hip/hip_runtime.h>
#include <hip/hip_bf16.h>

typedef __attribute__((ext_vector_type(16))) __bf16 v16bf;
typedef __attribute__((ext_vector_type(4)))  __bf16 v4bf;
typedef __attribute__((ext_vector_type(8)))  float  v8f;

#define N_TOKENS   (16 * 4096)   // 65536 z vectors
#define EMBED_DIM  64
#define N_EMBED    8192
#define CHUNK      256           // codebook rows staged in LDS per iteration
#define ROWS_PER_BLOCK 128       // 8 waves * 16 rows
#define BETA       0.25f

__global__ __launch_bounds__(256)
void vq_wmma_kernel(const float* __restrict__ z,
                    const float* __restrict__ emb,
                    float* __restrict__ zq,
                    float* __restrict__ lossp,
                    float* __restrict__ idxp)
{
    // bf16 codebook chunk + per-row squared norms
    __shared__ __bf16 ebf[CHUNK * EMBED_DIM];   // 32 KB
    __shared__ float  enorm[CHUNK];             // 1 KB

    const int tid  = threadIdx.x;
    const int lane = tid & 31;
    const int wave = tid >> 5;
    const int n    = lane & 15;      // column within 16-wide tile (B/C layout)
    const int h    = lane >> 4;      // lane half (K-group / M-group select)

    const int rowBase = blockIdx.x * ROWS_PER_BLOCK + wave * 16;
    const int myRow   = rowBase + n;

    // ---- Load this wave's 16 z rows as two bf16 A-fragments (16x32 each) ----
    // A-layout (16-bit, 16x32), matched to B: lanes 0-15 hold K = 0..15,
    // lanes 16-31 hold K = 16..31 (contiguous). Element j of v16bf -> VGPR j/2.
    const float* zrow = z + (size_t)myRow * EMBED_DIM;
    v16bf a0, a1;
    const int koff = h * 16;
#pragma unroll
    for (int j = 0; j < 16; ++j) {
        const int k0 = koff + j;
        a0[j] = (__bf16)zrow[k0];
        a1[j] = (__bf16)zrow[k0 + 32];
    }

    // Argmin state. NOTE: ||z_row||^2 is constant across the argmin dimension,
    // so the score is just  s = ||e||^2 - 2 * (z . e)  -- no znorm needed.
    float bestd[8];
    int   besti[8];
#pragma unroll
    for (int v = 0; v < 8; ++v) { bestd[v] = 3.0e38f; besti[v] = 0; }

    // ---- Sweep the codebook in LDS-staged chunks ----
    for (int cb = 0; cb < N_EMBED; cb += CHUNK) {
        __syncthreads();   // previous chunk's LDS reads complete

        // Stage chunk: fp32 -> bf16 conversion + row norms. One row per thread.
        {
            const int e = tid;                       // CHUNK == blockDim.x
            const float4* src = (const float4*)(emb + (size_t)(cb + e) * EMBED_DIM);
            v4bf* dst = (v4bf*)(ebf + e * EMBED_DIM);
            float s = 0.0f;
#pragma unroll
            for (int q = 0; q < 16; ++q) {
                const float4 f = src[q];
                s += f.x * f.x + f.y * f.y + f.z * f.z + f.w * f.w;
                v4bf b;
                b.x = (__bf16)f.x; b.y = (__bf16)f.y;
                b.z = (__bf16)f.z; b.w = (__bf16)f.w;
                dst[q] = b;
            }
            enorm[e] = s;
            // hint the next chunk toward L2/WGP$ (global_prefetch_b8)
            if (cb + CHUNK < N_EMBED)
                __builtin_prefetch(emb + (size_t)(cb + CHUNK + e) * EMBED_DIM, 0, 1);
        }
        __syncthreads();

        // 16 tiles of 16 codebook rows; 2 WMMAs per tile (K = 64).
        // Unroll x2: two independent WMMA chains in flight hides XDL->VALU
        // hazard NOPs and LDS latency.
#pragma unroll 2
        for (int nt = 0; nt < CHUNK / 16; ++nt) {
            const int entry = nt * 16 + n;
            // B-layout (16-bit, 32x16): lane n = column n; half h holds
            // K=16h..16h+15: 16 *contiguous* bf16 of codebook row `entry`.
            const __bf16* erow = ebf + entry * EMBED_DIM + h * 16;
            const v16bf b0 = *(const v16bf*)(erow);        // K 0..31 slice
            const v16bf b1 = *(const v16bf*)(erow + 32);   // K 32..63 slice

            v8f c = {};
            c = __builtin_amdgcn_wmma_f32_16x16x32_bf16(
                    false, a0, false, b0, (short)0, c, false, false);
            c = __builtin_amdgcn_wmma_f32_16x16x32_bf16(
                    false, a1, false, b1, (short)0, c, false, false);

            const float en   = enorm[entry];
            const int   gidx = cb + nt * 16 + n;
#pragma unroll
            for (int v = 0; v < 8; ++v) {
                const float d = __builtin_fmaf(-2.0f, c[v], en);
                if (d < bestd[v]) { bestd[v] = d; besti[v] = gidx; }
            }
        }
    }

    // ---- Argmin reduction across the 16 lanes of each half ----
    // (C/D layout: lane half h holds rows M = v + 8h; halves are disjoint,
    //  xor masks < 16 stay within a half)
#pragma unroll
    for (int v = 0; v < 8; ++v) {
        float d = bestd[v];
        int   bi = besti[v];
#pragma unroll
        for (int m = 1; m < 16; m <<= 1) {
            const float od = __shfl_xor(d, m, 32);
            const int   oi = __shfl_xor(bi, m, 32);
            if (od < d || (od == d && oi < bi)) { d = od; bi = oi; }
        }
        bestd[v] = d;
        besti[v] = bi;
    }

    // ---- Emit indices, gather z_q, accumulate loss ----
    float lossAcc = 0.0f;
#pragma unroll
    for (int v = 0; v < 8; ++v) {
        const int row = rowBase + v + 8 * h;
        const int bi  = besti[v];
        if (n == 0) idxp[row] = (float)bi;

        const float* er = emb + (size_t)bi * EMBED_DIM;
        const float* zr = z   + (size_t)row * EMBED_DIM;
        float*       qr = zq  + (size_t)row * EMBED_DIM;
#pragma unroll
        for (int j = 0; j < 4; ++j) {
            const int col = n * 4 + j;   // 16 lanes x 4 cols = 64
            const float e  = er[col];
            const float zz = zr[col];
            qr[col] = e;                 // straight-through: z + sg(zq - z) == zq
            const float dd = e - zz;
            lossAcc += dd * dd;
        }
    }

    // wave-wide loss reduction, then one atomic per wave
#pragma unroll
    for (int m = 1; m < 32; m <<= 1)
        lossAcc += __shfl_xor(lossAcc, m, 32);
    if (lane == 0) {
        const float scale = (1.0f + BETA) / (float)((size_t)N_TOKENS * EMBED_DIM);
        atomicAdd(lossp, lossAcc * scale);
    }
}

extern "C" void kernel_launch(void* const* d_in, const int* in_sizes, int n_in,
                              void* d_out, int out_size, void* d_ws, size_t ws_size,
                              hipStream_t stream)
{
    const float* z   = (const float*)d_in[0];   // [16,4096,64] fp32
    const float* emb = (const float*)d_in[1];   // [8192,64]    fp32

    float* out   = (float*)d_out;
    float* zq    = out;                                   // 16*4096*64
    float* lossp = out + (size_t)N_TOKENS * EMBED_DIM;    // 1
    float* idxp  = lossp + 1;                             // 16*4096 (as float)

    hipMemsetAsync(lossp, 0, sizeof(float), stream);      // capture-safe

    dim3 grid(N_TOKENS / ROWS_PER_BLOCK);  // 512 workgroups
    dim3 block(256);                       // 8 wave32s
    vq_wmma_kernel<<<grid, block, 0, stream>>>(z, emb, zq, lossp, idxp);
}
